// PerceiverAttentionLayer_2482491097557
// MI455X (gfx1250) — matrix operand order repaired and checked by reading, not compiled, not run on hardware
//
#include <hip/hip_runtime.h>
#include <hip/hip_bf16.h>

// ---------------------------------------------------------------------------
// Perceiver cross-attention for MI455X (gfx1250, wave32, WMMA).
// All GEMMs run on v_wmma_f32_16x16x32_bf16. Operands are pre-swizzled into
// the ISA fragment layouts so inner loops are pure 32B/lane global loads
// feeding WMMA; the fragment stream is ping-pong double-buffered (k-loop
// unrolled x2) so no register copies or full-drain waits appear in the loop.
// Activation set is L2-resident (192MB L2). Attention is flash-style with a
// 4-way kv split merged by log-sum-exp for occupancy.
// ---------------------------------------------------------------------------

typedef __bf16 bf16_t;
typedef __attribute__((ext_vector_type(16))) __bf16 v16bf;
typedef __attribute__((ext_vector_type(8)))  __bf16 v8bf;
typedef __attribute__((ext_vector_type(8)))  float  v8f;

#define HEADS  16
#define DIMH   64
#define NB     8
#define NF     4096
#define NQ     64
#define ND     1024
#define NFL    4160               // NF + NQ
#define ROWS_KV (NB * NFL)        // 33280 rows, 2080 row-tiles of 16
#define KT     32                 // K tiles of 32 over K=1024
#define FRAG   512                // elements per 32x16-tile fragment (32 lanes x 16)
#define SOFT_SCALE 0.125f         // 64^-0.5
#define FSPLIT 4
#define FCHUNKS 130               // NFL / 32
#define CPS    33                 // ceil(130/4) chunks per split

static __device__ __forceinline__ v8f wmma_bf16(v16bf a, v16bf b, v8f c) {
  return __builtin_amdgcn_wmma_f32_16x16x32_bf16(false, a, false, b, (short)0, c,
                                                 false, false);
}

// A-operand fragment index (16x32 bf16 tiles, K=1024 -> 32 k-chunks).
// Lane L (hs=L/16) holds row m=L%16, K elements {hs*8..hs*8+7, 16+hs*8..+7}.
static __device__ __forceinline__ size_t afrag_idx32(int rt, int m, int k) {
  int kc = k >> 5, r = k & 31;
  int hs = (r >> 3) & 1;
  int e  = (r & 7) | ((r >> 4) << 3);
  return (((size_t)rt * KT + kc) * 32 + (hs * 16 + m)) * 16 + e;
}

// ------------------------- weight prep: f32 -> bf16 B-frags -----------------
// W is KxN (1024x1024) row-major. B-frag: lane L holds column n=L%16,
// K elements {hs*8..hs*8+7, 16+hs*8..+7} (hs=L/16) within a 32-deep k-chunk. Layout: [nt][kt][lane][16].
__global__ __launch_bounds__(256)
void wprep_kernel(const float* __restrict__ W, bf16_t* __restrict__ dst) {
  int id = blockIdx.x * 256 + threadIdx.x;       // 1M threads
  int n = id & 1023, k = id >> 10;
  float v = W[(size_t)k * 1024 + n];
  int nt = n >> 4, kt = k >> 5;
  int L  = (((k >> 3) & 1) << 4) | (n & 15);
  int e  = (k & 7) | (((k >> 4) & 1) << 3);
  dst[(((size_t)nt * KT + kt) * 32 + L) * 16 + e] = (bf16_t)v;
}

// --------------- LayerNorm (features ++ latents) -> A-frag bf16 -------------
__global__ __launch_bounds__(256)
void ln_kernel(const float* __restrict__ features, const float* __restrict__ latents,
               const float* __restrict__ lnm_w, const float* __restrict__ lnm_b,
               const float* __restrict__ lnl_w, const float* __restrict__ lnl_b,
               bf16_t* __restrict__ xn) {
  int row = blockIdx.x;                           // 0..33279 (kv_in row)
  int b = row / NFL, rr = row - b * NFL;
  const float *src, *w, *bb;
  if (rr < NF) { src = features + ((size_t)b * NF + rr) * ND; w = lnm_w; bb = lnm_b; }
  else         { src = latents  + ((size_t)b * NQ + (rr - NF)) * ND; w = lnl_w; bb = lnl_b; }

  int tid = threadIdx.x;
  const float4 x = *reinterpret_cast<const float4*>(src + tid * 4);
  float s  = x.x + x.y + x.z + x.w;
  float s2 = x.x * x.x + x.y * x.y + x.z * x.z + x.w * x.w;
#pragma unroll
  for (int msk = 1; msk < 32; msk <<= 1) {
    s  += __shfl_xor(s,  msk, 32);
    s2 += __shfl_xor(s2, msk, 32);
  }
  __shared__ float red[2][8];
  int wave = tid >> 5, lane = tid & 31;
  if (lane == 0) { red[0][wave] = s; red[1][wave] = s2; }
  __syncthreads();
  float ts = 0.f, ts2 = 0.f;
#pragma unroll
  for (int i = 0; i < 8; ++i) { ts += red[0][i]; ts2 += red[1][i]; }
  float mu   = ts * (1.0f / ND);
  float var  = ts2 * (1.0f / ND) - mu * mu;
  float rstd = rsqrtf(var + 1e-5f);

  int rt = row >> 4, m = row & 15, k0 = tid * 4;
  float vx[4] = {x.x, x.y, x.z, x.w};
#pragma unroll
  for (int i = 0; i < 4; ++i) {
    int k = k0 + i;
    float y = (vx[i] - mu) * rstd * w[k] + bb[k];
    xn[afrag_idx32(rt, m, k)] = (bf16_t)y;
  }
}

// ------------------------- WMMA projection GEMM -----------------------------
// 8 waves/block; wave = 32 rows (2 A-frags) x 64 cols (4 B-frags) = 8 accums.
// k-loop unrolled x2 with ping-pong fragment buffers: no register copies,
// loads for step k+1 issued while step k's WMMAs execute.
// mode 0: KV rows -> K  frags [b][h][f/16][d/32]   (B-operand of QK^T)
// mode 1: KV rows -> V^T frags [b][h][f/32][d/16]  (B-operand of PV)
// mode 2: latent rows -> Q A-frags (softmax scale folded)
// mode 3: attention out rows -> f32 d_out
__global__ __launch_bounds__(256)
void gemm_kernel(const bf16_t* __restrict__ A, const bf16_t* __restrict__ Bw,
                 int mode, bf16_t* __restrict__ outb, float* __restrict__ outf,
                 float scale) {
  int tid = threadIdx.x, wave = tid >> 5, lane = tid & 31;
  int slot = blockIdx.x * 8 + wave;               // wave handles M-tiles 2slot,2slot+1
  int t0 = slot * 2, t1 = t0 + 1;
  int rt0 = (mode == 2) ? ((t0 >> 2) * 260 + 256 + (t0 & 3)) : t0;
  int rt1 = (mode == 2) ? ((t1 >> 2) * 260 + 256 + (t1 & 3)) : t1;
  int nb  = blockIdx.y;                           // 64-col slab (4 n-tiles)

  v8f zero = {0.f, 0.f, 0.f, 0.f, 0.f, 0.f, 0.f, 0.f};
  v8f acc[8];
#pragma unroll
  for (int j = 0; j < 8; ++j) acc[j] = zero;

  const bf16_t* a0p = A + ((size_t)rt0 * KT * 32 + lane) * 16;
  const bf16_t* a1p = A + ((size_t)rt1 * KT * 32 + lane) * 16;
  const bf16_t* bb  = Bw + ((size_t)(nb * 4) * KT * 32 + lane) * 16;

  // ping buffers (current)
  v16bf a0c = *reinterpret_cast<const v16bf*>(a0p);
  v16bf a1c = *reinterpret_cast<const v16bf*>(a1p);
  v16bf bc[4];
#pragma unroll
  for (int j = 0; j < 4; ++j)
    bc[j] = *reinterpret_cast<const v16bf*>(bb + (size_t)j * KT * FRAG);

  for (int kt = 0; kt < KT; kt += 2) {
    // --- even step: prefetch kt+1 into pong, compute on ping ---
    int k1 = kt + 1;                              // KT even -> always valid
    v16bf a0n = *reinterpret_cast<const v16bf*>(a0p + (size_t)k1 * FRAG);
    v16bf a1n = *reinterpret_cast<const v16bf*>(a1p + (size_t)k1 * FRAG);
    v16bf bn[4];
#pragma unroll
    for (int j = 0; j < 4; ++j)
      bn[j] = *reinterpret_cast<const v16bf*>(bb + ((size_t)j * KT + k1) * FRAG);
#pragma unroll
    for (int j = 0; j < 4; ++j) {
      acc[j]     = wmma_bf16(a0c, bc[j], acc[j]);
      acc[4 + j] = wmma_bf16(a1c, bc[j], acc[4 + j]);
    }
    // --- odd step: prefetch kt+2 into ping, compute on pong ---
    int k2 = (kt + 2 < KT) ? kt + 2 : KT - 1;     // clamped dummy on last iter
    a0c = *reinterpret_cast<const v16bf*>(a0p + (size_t)k2 * FRAG);
    a1c = *reinterpret_cast<const v16bf*>(a1p + (size_t)k2 * FRAG);
#pragma unroll
    for (int j = 0; j < 4; ++j)
      bc[j] = *reinterpret_cast<const v16bf*>(bb + ((size_t)j * KT + k2) * FRAG);
#pragma unroll
    for (int j = 0; j < 4; ++j) {
      acc[j]     = wmma_bf16(a0n, bn[j], acc[j]);
      acc[4 + j] = wmma_bf16(a1n, bn[j], acc[4 + j]);
    }
  }

  int hl = lane >> 4, nl = lane & 15;
#pragma unroll
  for (int i = 0; i < 2; ++i) {
    int t = t0 + i;
#pragma unroll
    for (int j = 0; j < 4; ++j) {
      int n = nb * 64 + j * 16 + nl;
#pragma unroll
      for (int v = 0; v < 8; ++v) {
        int   m   = v + 8 * hl;
        int   row = t * 16 + m;
        float val = acc[i * 4 + j][v] * scale;
        if (mode == 3) {
          outf[(size_t)row * ND + n] = val;
        } else if (mode == 0) {                   // K frags
          int b = row / NFL, f = row - b * NFL;
          int h = n >> 6, d = n & 63;
          size_t idx = ((((size_t)(b * 16 + h) * 260 + (f >> 4)) * 2 + (d >> 5)) * 32
                        + ((((d >> 3) & 1) << 4) | (f & 15))) * 16 + ((d & 7) | (((d >> 4) & 1) << 3));
          outb[idx] = (bf16_t)val;
        } else if (mode == 1) {                   // V^T frags
          int b = row / NFL, f = row - b * NFL;
          int h = n >> 6, d = n & 63;
          size_t idx = ((((size_t)(b * 16 + h) * 130 + (f >> 5)) * 4 + (d >> 4)) * 32
                        + ((((f >> 4) & 1) << 4) | (d & 15))) * 16 + (f & 15);
          outb[idx] = (bf16_t)val;
        } else {                                  // mode 2: Q A-frags
          int b = row >> 6, q = row & 63;
          int h = n >> 6, d = n & 63;
          int r32 = d & 31;
          int hs  = (r32 >> 3) & 1;
          int e   = (r32 & 7) | ((r32 >> 4) << 3);
          size_t idx = ((((size_t)(b * 16 + h) * 4 + (q >> 4)) * 2 + (d >> 5)) * 32
                        + (hs * 16 + (q & 15))) * 16 + e;
          outb[idx] = (bf16_t)val;
        }
      }
    }
  }
}

// ------------- flash attention, kv-split: block = (b,h) x split -------------
// Produces partial O (un-normalized), row-max m and row-sum l per split.
__global__ __launch_bounds__(128)
void attn_kernel(const bf16_t* __restrict__ qf, const bf16_t* __restrict__ kf,
                 const bf16_t* __restrict__ vtf, float* __restrict__ po,
                 float* __restrict__ pm, float* __restrict__ pl) {
  int bh  = blockIdx.x;                           // b*16 + h
  int sp  = blockIdx.y;                           // kv split 0..3
  int tid = threadIdx.x, w = tid >> 5, lane = tid & 31;
  int hl  = lane >> 4, nl = lane & 15;
  __shared__ bf16_t lds_p[4][16][40];             // 80B row stride: bank-clean

  const bf16_t* qb = qf + (((size_t)bh * 4 + w) * 2 * 32 + lane) * 16;
  v16bf qa0 = *reinterpret_cast<const v16bf*>(qb);
  v16bf qa1 = *reinterpret_cast<const v16bf*>(qb + FRAG);

  const bf16_t* kb = kf  + (size_t)bh * 260 * 2 * FRAG + (size_t)lane * 16;
  const bf16_t* vb = vtf + (size_t)bh * 130 * 4 * FRAG + (size_t)lane * 16;

  v8f zero = {0.f, 0.f, 0.f, 0.f, 0.f, 0.f, 0.f, 0.f};
  v8f o[4];
  float mrow[8], lrow[8];
#pragma unroll
  for (int i = 0; i < 4; ++i) o[i] = zero;
#pragma unroll
  for (int v = 0; v < 8; ++v) { mrow[v] = -3.0e38f; lrow[v] = 0.f; }

  int fc_beg = sp * CPS;
  int fc_end = (fc_beg + CPS < FCHUNKS) ? fc_beg + CPS : FCHUNKS;

  for (int fc = fc_beg; fc < fc_end; ++fc) {      // 32 kv rows per step
    v8f s0 = zero, s1 = zero;
    const bf16_t* k0 = kb + (size_t)(fc * 2 + 0) * 2 * FRAG;
    const bf16_t* k1 = kb + (size_t)(fc * 2 + 1) * 2 * FRAG;
    v16bf b00 = *reinterpret_cast<const v16bf*>(k0);
    v16bf b01 = *reinterpret_cast<const v16bf*>(k0 + FRAG);
    v16bf b10 = *reinterpret_cast<const v16bf*>(k1);
    v16bf b11 = *reinterpret_cast<const v16bf*>(k1 + FRAG);
    if (fc + 1 < fc_end) {
      __builtin_prefetch(kb + (size_t)((fc + 1) * 2) * 2 * FRAG, 0, 3);
      __builtin_prefetch(vb + (size_t)((fc + 1) * 4) * FRAG, 0, 3);
    }
    s0 = wmma_bf16(qa0, b00, s0);  s0 = wmma_bf16(qa1, b01, s0);
    s1 = wmma_bf16(qa0, b10, s1);  s1 = wmma_bf16(qa1, b11, s1);

    // online softmax; row r lives in one 16-lane half -> xor<16 reductions
#pragma unroll
    for (int v = 0; v < 8; ++v) {
      float c = fmaxf(s0[v], s1[v]);
      c = fmaxf(c, __shfl_xor(c, 1, 32));
      c = fmaxf(c, __shfl_xor(c, 2, 32));
      c = fmaxf(c, __shfl_xor(c, 4, 32));
      c = fmaxf(c, __shfl_xor(c, 8, 32));
      float mn   = fmaxf(mrow[v], c);
      float corr = __expf(mrow[v] - mn);
      float p0   = __expf(s0[v] - mn);
      float p1   = __expf(s1[v] - mn);
      float rs   = p0 + p1;
      rs += __shfl_xor(rs, 1, 32);
      rs += __shfl_xor(rs, 2, 32);
      rs += __shfl_xor(rs, 4, 32);
      rs += __shfl_xor(rs, 8, 32);
      lrow[v] = lrow[v] * corr + rs;
      mrow[v] = mn;
      o[0][v] *= corr; o[1][v] *= corr; o[2][v] *= corr; o[3][v] *= corr;
      int r = v + 8 * hl;
      lds_p[w][r][nl]      = (bf16_t)p0;
      lds_p[w][r][16 + nl] = (bf16_t)p1;
    }
    __syncthreads();

    // re-read P as a 16x32 A-fragment
    v16bf pa;
    {
      const bf16_t* pr = &lds_p[w][nl][0];
      v8bf lo = *reinterpret_cast<const v8bf*>(pr + hl * 16);
      v8bf hi = *reinterpret_cast<const v8bf*>(pr + hl * 16 + 8);
#pragma unroll
      for (int i = 0; i < 8; ++i) { pa[i] = lo[i]; pa[i + 8] = hi[i]; }
    }
#pragma unroll
    for (int nt = 0; nt < 4; ++nt) {
      v16bf vv = *reinterpret_cast<const v16bf*>(vb + (size_t)(fc * 4 + nt) * FRAG);
      o[nt] = wmma_bf16(pa, vv, o[nt]);
    }
    __syncthreads();
  }

  // write partials (O un-normalized; per-row m, l)
  int sb = sp * 128 + bh;
  float* pob = po + (size_t)sb * 64 * 64;
#pragma unroll
  for (int nt = 0; nt < 4; ++nt) {
#pragma unroll
    for (int v = 0; v < 8; ++v) {
      int q = w * 16 + v + 8 * hl;
      int d = nt * 16 + nl;
      pob[(size_t)q * 64 + d] = o[nt][v];
    }
  }
  if (nl == 0) {
#pragma unroll
    for (int v = 0; v < 8; ++v) {
      int q = w * 16 + v + 8 * hl;
      pm[(size_t)sb * 64 + q] = mrow[v];
      pl[(size_t)sb * 64 + q] = lrow[v];
    }
  }
}

// --------- merge kv-splits (log-sum-exp) and scatter to A-frag bf16 ---------
__global__ __launch_bounds__(256)
void attn_merge_kernel(const float* __restrict__ po, const float* __restrict__ pm,
                       const float* __restrict__ pl, bf16_t* __restrict__ ao) {
  int id = blockIdx.x * 256 + threadIdx.x;        // 128*64*64 threads
  int d  = id & 63;
  int q  = (id >> 6) & 63;
  int bh = id >> 12;

  float ms[FSPLIT], M = -3.0e38f;
#pragma unroll
  for (int s = 0; s < FSPLIT; ++s) {
    ms[s] = pm[(size_t)(s * 128 + bh) * 64 + q];
    M = fmaxf(M, ms[s]);
  }
  float L = 0.f, ov = 0.f;
#pragma unroll
  for (int s = 0; s < FSPLIT; ++s) {
    float wgt = __expf(ms[s] - M);
    L  += pl[(size_t)(s * 128 + bh) * 64 + q] * wgt;
    ov += po[((size_t)(s * 128 + bh) * 64 + q) * 64 + d] * wgt;
  }
  float res = ov / L;

  int b = bh >> 4, h = bh & 15;
  int rt = ((b << 6) + q) >> 4;                   // = b*4 + q/16
  int m  = q & 15;
  int k  = (h << 6) + d;
  ao[afrag_idx32(rt, m, k)] = (bf16_t)res;
}

// ---------------------------------------------------------------------------
extern "C" void kernel_launch(void* const* d_in, const int* in_sizes, int n_in,
                              void* d_out, int out_size, void* d_ws, size_t ws_size,
                              hipStream_t stream) {
  (void)in_sizes; (void)n_in; (void)out_size; (void)ws_size;
  const float* features = (const float*)d_in[0];
  const float* latents  = (const float*)d_in[1];
  const float* lnm_w    = (const float*)d_in[2];
  const float* lnm_b    = (const float*)d_in[3];
  const float* lnl_w    = (const float*)d_in[4];
  const float* lnl_b    = (const float*)d_in[5];
  const float* Wq       = (const float*)d_in[6];
  const float* Wk       = (const float*)d_in[7];
  const float* Wv       = (const float*)d_in[8];
  const float* Wo       = (const float*)d_in[9];
  float* out = (float*)d_out;

  char* ws = (char*)d_ws;
  size_t off = 0;
  auto alloc = [&](size_t bytes) {
    char* p = ws + off;
    off += (bytes + 255) & ~(size_t)255;
    return p;
  };
  bf16_t* WQf = (bf16_t*)alloc((size_t)2 << 20);
  bf16_t* WKf = (bf16_t*)alloc((size_t)2 << 20);
  bf16_t* WVf = (bf16_t*)alloc((size_t)2 << 20);
  bf16_t* WOf = (bf16_t*)alloc((size_t)2 << 20);
  bf16_t* xn  = (bf16_t*)alloc((size_t)ROWS_KV * 1024 * 2);          // 68.2 MB
  bf16_t* kfb = (bf16_t*)alloc((size_t)128 * 260 * 2 * FRAG * 2);    // 68.2 MB
  bf16_t* vtf = (bf16_t*)alloc((size_t)128 * 130 * 4 * FRAG * 2);    // 68.2 MB
  bf16_t* qfb = (bf16_t*)alloc((size_t)128 * 4 * 2 * FRAG * 2);      // 1 MB
  bf16_t* aof = (bf16_t*)alloc((size_t)512 * 1024 * 2);              // 1 MB
  float*  po  = (float*)alloc((size_t)FSPLIT * 128 * 64 * 64 * 4);   // 8.4 MB
  float*  pm  = (float*)alloc((size_t)FSPLIT * 128 * 64 * 4);
  float*  pl  = (float*)alloc((size_t)FSPLIT * 128 * 64 * 4);

  wprep_kernel<<<4096, 256, 0, stream>>>(Wq, WQf);
  wprep_kernel<<<4096, 256, 0, stream>>>(Wk, WKf);
  wprep_kernel<<<4096, 256, 0, stream>>>(Wv, WVf);
  wprep_kernel<<<4096, 256, 0, stream>>>(Wo, WOf);

  ln_kernel<<<ROWS_KV, 256, 0, stream>>>(features, latents, lnm_w, lnm_b,
                                         lnl_w, lnl_b, xn);

  gemm_kernel<<<dim3(130, 16), 256, 0, stream>>>(xn, WKf, 0, kfb, nullptr, 1.0f);
  gemm_kernel<<<dim3(130, 16), 256, 0, stream>>>(xn, WVf, 1, vtf, nullptr, 1.0f);
  gemm_kernel<<<dim3(2, 16),   256, 0, stream>>>(xn, WQf, 2, qfb, nullptr, SOFT_SCALE);

  attn_kernel<<<dim3(128, FSPLIT), 128, 0, stream>>>(qfb, kfb, vtf, po, pm, pl);
  attn_merge_kernel<<<2048, 256, 0, stream>>>(po, pm, pl, aof);

  gemm_kernel<<<dim3(2, 16),   256, 0, stream>>>(aof, WOf, 3, nullptr, out, 1.0f);
}
